// KANMAMMOTE_21904333209602
// MI455X (gfx1250) — compile-verified, matrix operand order checked by reading, not run on hardware
//
#include <hip/hip_runtime.h>

// ---------------------------------------------------------------------------
// KAN-MAMMOTE forward for MI455X (gfx1250, wave32, WMMA).
// GEMMs: bf16 WMMA (v_wmma_f32_16x16x32_bf16) with f32 accumulation.
// LDS tiles are staged PRE-SWIZZLED into WMMA fragment order so each lane
// loads its 16-element fragment with one contiguous 32-byte LDS read
// (ds_load_b128 pair) instead of 16 ds_load_u16 + v_mov_b16 merges.
// ---------------------------------------------------------------------------

typedef __attribute__((ext_vector_type(16))) __bf16 v16bf;
typedef __attribute__((ext_vector_type(8)))  float  v8f;

#define BB     8
#define LL     2048
#define FF     128
#define DD     256
#define NLAYER 4
#define DTE    64
#define KEXP   8
#define TOPKN  2
#define GG     8
#define DI     512
#define DS     16
#define CWW    4
#define OUTD   128
#define MROWS  (BB * LL)

static __device__ __forceinline__ unsigned short f2bf(float f) {
  // round-to-nearest-even f32 -> bf16 (bit-level)
  unsigned int u = __float_as_uint(f);
  unsigned int r = u + 0x7FFFu + ((u >> 16) & 1u);
  return (unsigned short)(r >> 16);
}

static __device__ __forceinline__ unsigned long long pack4bf(float a, float b,
                                                            float c, float d) {
  return (unsigned long long)f2bf(a) | ((unsigned long long)f2bf(b) << 16) |
         ((unsigned long long)f2bf(c) << 32) | ((unsigned long long)f2bf(d) << 48);
}

// ---------------------------------------------------------------------------
// Generic tiled GEMM: C[M,N] = epilogue(A[M,K] @ W[K,N]).
// Block = 128 threads (4 waves). Block tile 64x64, K-chunk 64 (2 WMMA steps).
// LDS layout: fragment order [kc(2)][subtile(4)][lane(32)][elem(16)] bf16.
//   A fragment mapping (16-bit 16x32):  lane = (row&15) | khalf<<4,
//     khalf=(k>>3)&1, kb=k-8*khalf, elem i0 = kb<8 ? kb : kb-8 (4-aligned runs
//     of k are 4 consecutive elems -> staged with one 8-byte LDS store).
//   B mapping is the transpose-symmetric layout (lane carries the column).
// Wave w owns rows [w*16,w*16+16); 4 accumulators cover 64 columns.
// Epilogue: optional bias, softplus (act==1), residual add (Cres).
// All call sites: M%64==0, K%64==0; N guarded (N=32 case).
// ---------------------------------------------------------------------------
__global__ __launch_bounds__(128) void gemm_bf16_wmma(
    const float* __restrict__ A, const float* __restrict__ W,
    const float* __restrict__ bias, const float* __restrict__ Cres,
    float* __restrict__ C, int M, int N, int Kd, int act)
{
  __shared__ __align__(32) unsigned short lA[2 * 4 * 32 * 16];  // 8 KB
  __shared__ __align__(32) unsigned short lB[2 * 4 * 32 * 16];  // 8 KB
  const int tid  = threadIdx.x;
  const int wave = tid >> 5;
  const int lane = tid & 31;
  const int bm   = blockIdx.y * 64;
  const int bn   = blockIdx.x * 64;

  v8f acc[4];
#pragma unroll
  for (int i = 0; i < 4; ++i) acc[i] = (v8f){0, 0, 0, 0, 0, 0, 0, 0};

  for (int k0 = 0; k0 < Kd; k0 += 64) {
    // ---- stage A tile (64 rows x 64 k): float4 load -> one b64 LDS store ----
#pragma unroll
    for (int it = 0; it < 8; ++it) {
      int idx = tid + it * 128;          // 1024 tasks
      int r   = idx >> 4;                // row in tile
      int kq  = (idx & 15) << 2;         // 4-aligned k in tile
      const float4 f =
          *reinterpret_cast<const float4*>(A + (size_t)(bm + r) * Kd + k0 + kq);
      int kc    = kq >> 5;
      int k     = kq & 31;
      int khalf = (k >> 3) & 1;
      int kb    = k - (khalf << 3);
      int i0    = (kb < 8) ? kb : (kb - 8);
      int ln    = (r & 15) | (khalf << 4);
      int mt    = r >> 4;
      int dst   = (((kc << 2) | mt) * 32 + ln) * 16 + i0;
      *reinterpret_cast<unsigned long long*>(&lA[dst]) = pack4bf(f.x, f.y, f.z, f.w);
    }
    // ---- stage B tile (64 k x 64 n): 4 coalesced loads down K -> b64 store ----
#pragma unroll
    for (int it = 0; it < 8; ++it) {
      int idx = tid + it * 128;          // 1024 tasks
      int c   = idx & 63;                // column in tile
      int kq  = (idx >> 6) << 2;         // 4-aligned k in tile
      int col = bn + c;
      float v0 = 0.f, v1 = 0.f, v2 = 0.f, v3 = 0.f;
      if (col < N) {
        const float* bp = W + (size_t)(k0 + kq) * N + col;
        v0 = bp[0]; v1 = bp[N]; v2 = bp[2 * (size_t)N]; v3 = bp[3 * (size_t)N];
      }
      int kc    = kq >> 5;
      int k     = kq & 31;
      int khalf = (k >> 3) & 1;
      int kb    = k - (khalf << 3);
      int i0    = (kb < 8) ? kb : (kb - 8);
      int ln    = (c & 15) | (khalf << 4);
      int nbt   = c >> 4;
      int dst   = (((kc << 2) | nbt) * 32 + ln) * 16 + i0;
      *reinterpret_cast<unsigned long long*>(&lB[dst]) = pack4bf(v0, v1, v2, v3);
    }
    __syncthreads();

    // ---- compute: fragments are contiguous 32-byte LDS reads ----
#pragma unroll
    for (int kc = 0; kc < 2; ++kc) {
      const v16bf av = *reinterpret_cast<const v16bf*>(
          &lA[(((kc << 2) | wave) * 32 + lane) * 16]);
#pragma unroll
      for (int nb = 0; nb < 4; ++nb) {
        const v16bf bv = *reinterpret_cast<const v16bf*>(
            &lB[(((kc << 2) | nb) * 32 + lane) * 16]);
        acc[nb] = __builtin_amdgcn_wmma_f32_16x16x32_bf16(
            false, av, false, bv, (short)0, acc[nb], false, false);
      }
    }
    __syncthreads();
  }

  // C/D layout: VGPR r, lane -> m = r + 8*(lane>>4), n = lane&15
  const int row16 = lane & 15;
  const int mo    = (lane >> 4) << 3;
#pragma unroll
  for (int nb = 0; nb < 4; ++nb) {
    int col = bn + nb * 16 + row16;
    if (col >= N) continue;
    float bv = bias ? bias[col] : 0.0f;
#pragma unroll
    for (int r = 0; r < 8; ++r) {
      int m   = bm + wave * 16 + mo + r;
      float v = acc[nb][r] + bv;
      if (act == 1) v = (v > 20.0f) ? v : log1pf(__expf(v));  // softplus
      size_t off = (size_t)m * N + col;
      if (Cres) v += Cres[off];
      C[off] = v;
    }
  }
}

// ---------------------------------------------------------------------------
// Fused K-MoTE router + sinusoidal experts + FasterKAN delta embedding.
// One block (64 threads) per (b,t). KAN mix is linear so
//   dte = (basis(emb_t) - basis(emb_prev)) @ kan_W
// -> build the basis DIFFERENCE (512 values) in LDS, then one matvec.
// Threads 0..7 accumulate gate/selection sums for the LB loss.
// ---------------------------------------------------------------------------
__global__ __launch_bounds__(64) void kmote_kan_kernel(
    const float* __restrict__ ts, const float* __restrict__ freqs,
    const float* __restrict__ Wr, const float* __restrict__ brv,
    const float* __restrict__ grid, const float* __restrict__ kanW,
    float* __restrict__ dte, float* __restrict__ accum)
{
  __shared__ float bdiff[DTE * GG];
  const int n   = blockIdx.x;
  const int d   = threadIdx.x;
  const int pos = n & (LL - 1);
  const float t  = ts[n];
  const float tp = (pos == 0) ? t : ts[n - 1];

  auto embed = [&](float tv, bool do_stats) -> float {
    float lg[KEXP];
    float mx = -1e30f;
#pragma unroll
    for (int k = 0; k < KEXP; ++k) { lg[k] = tv * Wr[k] + brv[k]; mx = fmaxf(mx, lg[k]); }
    float se = 0.0f;
#pragma unroll
    for (int k = 0; k < KEXP; ++k) { lg[k] = __expf(lg[k] - mx); se += lg[k]; }
#pragma unroll
    for (int k = 0; k < KEXP; ++k) lg[k] /= se;
    // top-2 (lowest index wins ties, matching lax.top_k)
    int i1 = 0; float g1 = lg[0];
#pragma unroll
    for (int k = 1; k < KEXP; ++k) if (lg[k] > g1) { g1 = lg[k]; i1 = k; }
    int i2 = (i1 == 0) ? 1 : 0; float g2 = lg[i2];
#pragma unroll
    for (int k = 0; k < KEXP; ++k) if (k != i1 && lg[k] > g2) { g2 = lg[k]; i2 = k; }
    float wsum = g1 + g2 + 1e-9f;
    float w1 = g1 / wsum, w2 = g2 / wsum;
    if (do_stats && d < KEXP) {
      atomicAdd(&accum[d], lg[d]);                                     // gates
      atomicAdd(&accum[KEXP + d], (d == i1 || d == i2) ? 1.0f : 0.0f); // sel
    }
    float e;
    if (d < DTE / 2) {
      e = w1 * __sinf(tv * freqs[i1 * (DTE / 2) + d]) +
          w2 * __sinf(tv * freqs[i2 * (DTE / 2) + d]);
    } else {
      int dd = d - DTE / 2;
      e = w1 * __cosf(tv * freqs[i1 * (DTE / 2) + dd]) +
          w2 * __cosf(tv * freqs[i2 * (DTE / 2) + dd]);
    }
    return e;
  };

  const float emb_t = embed(t, true);
  const float emb_p = embed(tp, false);

  const float h = grid[1] - grid[0];
#pragma unroll
  for (int g = 0; g < GG; ++g) {
    float gv = grid[g];
    float a1 = tanhf((emb_t - gv) / h);
    float a2 = tanhf((emb_p - gv) / h);
    bdiff[d * GG + g] = (a2 * a2) - (a1 * a1);  // (1-a1^2)-(1-a2^2)
  }
  __syncthreads();

  float acc = 0.0f;
  for (int r = 0; r < DTE * GG; ++r) acc += bdiff[r] * kanW[(size_t)r * DTE + d];
  dte[(size_t)n * DTE + d] = acc;
}

// ---------------------------------------------------------------------------
// Causal depthwise conv (width 4) over the x-half of xz, fused with SiLU.
// ---------------------------------------------------------------------------
__global__ __launch_bounds__(256) void conv_silu_kernel(
    const float* __restrict__ xz, const float* __restrict__ cw,
    float* __restrict__ xc)
{
  size_t idx = (size_t)blockIdx.x * 256 + threadIdx.x;
  if (idx >= (size_t)MROWS * DI) return;
  int    di  = (int)(idx % DI);
  size_t row = idx / DI;            // b*L + t
  int    t   = (int)(row & (LL - 1));
  float acc = 0.0f;
#pragma unroll
  for (int k = 0; k < CWW; ++k) {
    int tt = t + k - (CWW - 1);
    if (tt >= 0)
      acc += xz[(row + (size_t)(long)(k - (CWW - 1))) * (2 * DI) + di] * cw[di * CWW + k];
  }
  acc = acc / (1.0f + __expf(-acc));  // SiLU
  xc[idx] = acc;
}

// ---------------------------------------------------------------------------
// Selective-scan: one thread per (b, di); DS=16 state in registers.
// B/C rows loaded as float4 (8 wide vmem loads per step).
// Fuses D-skip and SiLU(z) gating into the output write.
// ---------------------------------------------------------------------------
__global__ __launch_bounds__(256) void ssm_scan_kernel(
    const float* __restrict__ xc, const float* __restrict__ dtb,
    const float* __restrict__ bc, const float* __restrict__ xz,
    const float* __restrict__ A_log, const float* __restrict__ Dskip,
    float* __restrict__ y)
{
  int gid = blockIdx.x * 256 + threadIdx.x;   // 0 .. B*DI-1
  int b   = gid >> 9;                          // DI == 512
  int di  = gid & (DI - 1);
  float A  = -__expf(A_log[di]);
  float Dl = Dskip[di];
  float s[DS];
#pragma unroll
  for (int i = 0; i < DS; ++i) s[i] = 0.0f;

  for (int t = 0; t < LL; ++t) {
    size_t row = (size_t)b * LL + t;
    float xt  = xc[row * DI + di];
    float dtt = dtb[row * DI + di];
    float zt  = xz[row * (2 * DI) + DI + di];
    const float4* bc4 = reinterpret_cast<const float4*>(bc + row * (2 * DS));
    float Btl[DS], Ctl[DS];
#pragma unroll
    for (int q = 0; q < DS / 4; ++q) {
      *reinterpret_cast<float4*>(&Btl[q * 4]) = bc4[q];
      *reinterpret_cast<float4*>(&Ctl[q * 4]) = bc4[DS / 4 + q];
    }
    float e  = __expf(dtt * A);
    float cf = dtt * xt;
    float ya = 0.0f;
#pragma unroll
    for (int i = 0; i < DS; ++i) {
      s[i] = e * s[i] + cf * Btl[i];
      ya  += s[i] * Ctl[i];
    }
    float sz = zt / (1.0f + __expf(-zt));
    y[row * DI + di] = (ya + xt * Dl) * sz;
    if (t + 1 < LL) {
      __builtin_prefetch(xc + (row + 1) * DI + di, 0, 1);
      __builtin_prefetch(bc + (row + 1) * (2 * DS), 0, 1);
    }
  }
}

// ---------------------------------------------------------------------------
// Regularizer reductions over kan_W: sobolev L2 and total variation along G.
// ---------------------------------------------------------------------------
__global__ __launch_bounds__(256) void kan_reg_kernel(
    const float* __restrict__ kanW, float* __restrict__ accum)
{
  __shared__ float s1[256], s2[256];
  int i = blockIdx.x * 256 + threadIdx.x;
  float w = 0.0f, tvv = 0.0f;
  if (i < DTE * GG * DTE) {
    w = kanW[i];
    int g = (i / DTE) % GG;             // flat = (d*G + g)*DT + j
    if (g < GG - 1) tvv = fabsf(kanW[i + DTE] - w);
  }
  s1[threadIdx.x] = w * w;
  s2[threadIdx.x] = tvv;
  __syncthreads();
  for (int st = 128; st > 0; st >>= 1) {
    if (threadIdx.x < st) {
      s1[threadIdx.x] += s1[threadIdx.x + st];
      s2[threadIdx.x] += s2[threadIdx.x + st];
    }
    __syncthreads();
  }
  if (threadIdx.x == 0) {
    atomicAdd(&accum[16], s1[0]);
    atomicAdd(&accum[17], s2[0]);
  }
}

__global__ void init_accum_kernel(float* __restrict__ accum) {
  if (threadIdx.x < 32) accum[threadIdx.x] = 0.0f;
}

__global__ void finalize_kernel(const float* __restrict__ accum,
                                float* __restrict__ outaux) {
  if (threadIdx.x == 0) {
    const float invN = 1.0f / (float)MROWS;
    float lb = 0.0f;
#pragma unroll
    for (int k = 0; k < KEXP; ++k)
      lb += (accum[k] * invN) * (accum[KEXP + k] * invN) / (float)TOPKN;
    lb *= (float)KEXP;
    outaux[0] = lb;
    outaux[1] = accum[16];
    outaux[2] = accum[17];
  }
}

// ---------------------------------------------------------------------------
extern "C" void kernel_launch(void* const* d_in, const int* in_sizes, int n_in,
                              void* d_out, int out_size, void* d_ws, size_t ws_size,
                              hipStream_t stream)
{
  const float* ts     = (const float*)d_in[0];
  const float* feats  = (const float*)d_in[1];
  const float* freqs  = (const float*)d_in[2];
  const float* Wr     = (const float*)d_in[3];
  const float* brv    = (const float*)d_in[4];
  const float* grid   = (const float*)d_in[5];
  const float* kanW   = (const float*)d_in[6];
  const float* W_in   = (const float*)d_in[7];
  const float* b_in   = (const float*)d_in[8];
  const float* Win    = (const float*)d_in[9];
  const float* conv_w = (const float*)d_in[10];
  const float* Wdt    = (const float*)d_in[11];
  const float* bdt    = (const float*)d_in[12];
  const float* Wbc    = (const float*)d_in[13];
  const float* A_log  = (const float*)d_in[14];
  const float* Dskip  = (const float*)d_in[15];
  const float* Wout   = (const float*)d_in[16];
  const float* W_head = (const float*)d_in[17];
  const float* b_head = (const float*)d_in[18];
  float* out = (float*)d_out;

  float* ws = (float*)d_ws;
  size_t o = 0;
  float* xh    = ws + o; o += (size_t)MROWS * DD;        // residual stream h
  float* dte   = ws + o; o += (size_t)MROWS * DTE;       // delta-time embedding
  float* xz    = ws + o; o += (size_t)MROWS * 2 * DI;    // in-proj (x | z)
  float* xc    = ws + o; o += (size_t)MROWS * DI;        // conv+silu(x)
  float* dtb   = ws + o; o += (size_t)MROWS * DI;        // softplus dt
  float* bcb   = ws + o; o += (size_t)MROWS * 2 * DS;    // (B | C)
  float* yb    = ws + o; o += (size_t)MROWS * DI;        // gated scan output
  float* accum = ws + o; o += 32;                        // gate sums, sob, tv

  init_accum_kernel<<<1, 32, 0, stream>>>(accum);
  kmote_kan_kernel<<<MROWS, DTE, 0, stream>>>(ts, freqs, Wr, brv, grid, kanW,
                                              dte, accum);

  auto gemm = [&](const float* A, const float* W, const float* bias,
                  const float* Cres, float* C, int M, int N, int Kd, int act) {
    dim3 g((N + 63) / 64, M / 64);
    gemm_bf16_wmma<<<g, dim3(128), 0, stream>>>(A, W, bias, Cres, C, M, N, Kd, act);
  };

  // initial feature projection: [M,128] @ [128,256] + b
  gemm(feats, W_in, b_in, nullptr, xh, MROWS, DD, FF, 0);

  for (int l = 0; l < NLAYER; ++l) {
    // in-proj: [M,256] @ [256,1024]
    gemm(xh, Win + (size_t)l * DD * 2 * DI, nullptr, nullptr, xz,
         MROWS, 2 * DI, DD, 0);
    // causal depthwise conv + SiLU
    {
      size_t tot = (size_t)MROWS * DI;
      conv_silu_kernel<<<(unsigned)((tot + 255) / 256), 256, 0, stream>>>(
          xz, conv_w + (size_t)l * DI * CWW, xc);
    }
    // dt = softplus(dte @ Wdt + bdt): [M,64] @ [64,512]
    gemm(dte, Wdt + (size_t)l * DTE * DI, bdt + (size_t)l * DI, nullptr, dtb,
         MROWS, DI, DTE, 1);
    // B,C projection: [M,512] @ [512,32]
    gemm(xc, Wbc + (size_t)l * DI * 2 * DS, nullptr, nullptr, bcb,
         MROWS, 2 * DS, DI, 0);
    // sequential selective scan, fused skip + gating
    ssm_scan_kernel<<<(BB * DI) / 256, 256, 0, stream>>>(
        xc, dtb, bcb, xz, A_log + (size_t)l * DI, Dskip + (size_t)l * DI, yb);
    // out-proj + residual: h += y @ Wout  ([M,512] @ [512,256])
    gemm(yb, Wout + (size_t)l * DI * DD, nullptr, xh, xh, MROWS, DD, DI, 0);
  }

  // prediction head: [M,256] @ [256,128] + b -> d_out
  gemm(xh, W_head, b_head, nullptr, out, MROWS, OUTD, DD, 0);

  // aux losses -> d_out[M*OUT .. +3)
  kan_reg_kernel<<<(DTE * GG * DTE) / 256, 256, 0, stream>>>(kanW, accum);
  finalize_kernel<<<1, 32, 0, stream>>>(accum, out + (size_t)MROWS * OUTD);
}
